// SpatialLanguageModel_68513318306529
// MI455X (gfx1250) — matrix operand order repaired and checked by reading, not compiled, not run on hardware
//
#include <hip/hip_runtime.h>
#include <math.h>

// ---- problem constants (from reference) ----
#define BB 4
#define SS 1024
#define DD 768
#define HH 12
#define LL 4
#define VV 256
#define DHD 64
#define D3 (3*DD)
#define D4 (4*DD)
#define BT (BB*SS)

typedef __bf16 bf16_t;
typedef __attribute__((ext_vector_type(16))) __bf16 v16bf;
typedef __attribute__((ext_vector_type(8)))  __bf16 v8bf;
typedef __attribute__((ext_vector_type(8)))  float  v8f;

// ------------------------------------------------------------------
// DPP16 16-lane butterfly reductions (no LDS, no waits).
// masks {1,2,7,15} (quad_perm, quad_perm, row_half_mirror, row_mirror)
// form a basis of F2^4 -> every lane of each 16-lane group ends with
// the full group reduction; the two wave halves stay independent.
// ------------------------------------------------------------------
template <int CTRL>
__device__ __forceinline__ float dpp_f32(float v) {
  return __int_as_float(
      __builtin_amdgcn_mov_dpp(__float_as_int(v), CTRL, 0xF, 0xF, true));
}

__device__ __forceinline__ float rmax16(float v) {
  v = fmaxf(v, dpp_f32<0xB1>(v));   // quad_perm [1,0,3,2]  (xor 1)
  v = fmaxf(v, dpp_f32<0x4E>(v));   // quad_perm [2,3,0,1]  (xor 2)
  v = fmaxf(v, dpp_f32<0x141>(v));  // row_half_mirror      (xor 7)
  v = fmaxf(v, dpp_f32<0x140>(v));  // row_mirror           (xor 15)
  return v;
}

__device__ __forceinline__ float rsum16(float v) {
  v += dpp_f32<0xB1>(v);
  v += dpp_f32<0x4E>(v);
  v += dpp_f32<0x141>(v);
  v += dpp_f32<0x140>(v);
  return v;
}

// ------------------------------------------------------------------
// WMMA fragment helpers (layouts per CDNA5 ISA 7.12.2, wave32)
// ------------------------------------------------------------------
__device__ __forceinline__ v8f zero_v8f() {
  v8f r;
#pragma unroll
  for (int i = 0; i < 8; ++i) r[i] = 0.f;
  return r;
}

__device__ __forceinline__ v16bf pack16(v8bf lo, v8bf hi) {
  v16bf r;
#pragma unroll
  for (int i = 0; i < 8; ++i) { r[i] = lo[i]; r[8 + i] = hi[i]; }
  return r;
}

// A fragment: 16x32 bf16, row-major source with stride `stride` (elements).
// lanes 0-15: row m=lane, K in {0..7, 16..23}; lanes 16-31: row m=lane-16, K in {8..15, 24..31}
__device__ __forceinline__ v16bf load_a_frag(const bf16_t* __restrict__ base, int stride) {
  const int lane = threadIdx.x & 31;
  const int m    = lane & 15;
  const int kb   = (lane < 16) ? 0 : 8;
  const bf16_t* rp = base + (size_t)m * stride + kb;
  v8bf lo = *reinterpret_cast<const v8bf*>(rp);        // K kb..kb+7
  v8bf hi = *reinterpret_cast<const v8bf*>(rp + 16);   // K kb+16..kb+23
  return pack16(lo, hi);
}

// B fragment: 32x16 bf16 where B is stored TRANSPOSED (N x K row-major, stride ldbt).
// lanes 0-15: col n=lane, K in {0..7, 16..23}; lanes 16-31: col n=lane-16, K in {8..15, 24..31}
__device__ __forceinline__ v16bf load_bt_frag(const bf16_t* __restrict__ baseT, int strideT) {
  const int lane = threadIdx.x & 31;
  const int n    = lane & 15;
  const int kofs = (lane < 16) ? 0 : 8;
  const bf16_t* rp = baseT + (size_t)n * strideT + kofs;
  v8bf lo = *reinterpret_cast<const v8bf*>(rp);
  v8bf hi = *reinterpret_cast<const v8bf*>(rp + 16);
  return pack16(lo, hi);
}

__device__ __forceinline__ v8f wmma_bf16(v16bf a, v16bf b, v8f c) {
  return __builtin_amdgcn_wmma_f32_16x16x32_bf16(false, a, false, b, (short)0, c, false, false);
}

// ------------------------------------------------------------------
// small utility kernels
// ------------------------------------------------------------------
__global__ void zero_acc_kernel(float* acc) {
  if (threadIdx.x < 16) acc[threadIdx.x] = 0.f;
}

__global__ void cvt_bf16_kernel(const float* __restrict__ in, bf16_t* __restrict__ out, size_t n) {
  size_t i = (size_t)blockIdx.x * blockDim.x + threadIdx.x;
  if (i < n) out[i] = (bf16_t)in[i];
}

// convert f32 (K x N) -> bf16 transposed (N x K); batched over blockIdx.z layers
__global__ void cvt_t_bf16_kernel(const float* __restrict__ in, bf16_t* __restrict__ out,
                                  int K, int N) {
  size_t base = (size_t)blockIdx.z * K * N;
  size_t i = (size_t)blockIdx.x * blockDim.x + threadIdx.x;
  if (i < (size_t)K * N) {
    int k = (int)(i / N);
    int n = (int)(i % N);
    out[base + (size_t)n * K + k] = (bf16_t)in[base + i];
  }
}

// repack V slice of qkv into dh-major layout: vT[((b*H+h)*dh + n)*S + t]
__global__ void vtrans_kernel(const bf16_t* __restrict__ qkvb, bf16_t* __restrict__ vT) {
  size_t i = (size_t)blockIdx.x * blockDim.x + threadIdx.x;
  const size_t total = (size_t)BB * DD * SS;
  if (i >= total) return;
  int t    = (int)(i % SS);
  int rest = (int)(i / SS);     // b*DD + (h*DHD + n)
  int d    = rest % DD;         // h*DHD + n
  int b    = rest / DD;
  vT[i] = qkvb[((size_t)(b * SS + t)) * D3 + 2 * DD + d];
}

__global__ void embed_kernel(const int* __restrict__ idx, const float* __restrict__ te,
                             const float* __restrict__ pe, const float* __restrict__ tp,
                             const float* __restrict__ ppw, float* __restrict__ x,
                             float* __restrict__ sp) {
  int bt = blockIdx.x;
  int t  = bt % SS;
  int id = idx[bt];
  float s0 = tp[id * 3 + 0], s1 = tp[id * 3 + 1], s2 = tp[id * 3 + 2];
  if (threadIdx.x == 0) { sp[bt * 3 + 0] = s0; sp[bt * 3 + 1] = s1; sp[bt * 3 + 2] = s2; }
  for (int d = threadIdx.x; d < DD; d += blockDim.x) {
    x[(size_t)bt * DD + d] = te[(size_t)id * DD + d] + pe[(size_t)t * DD + d]
                           + s0 * ppw[d] + s1 * ppw[DD + d] + s2 * ppw[2 * DD + d];
  }
}

__global__ void dist_kernel(const float* __restrict__ sp, float* __restrict__ dist) {
  size_t i = (size_t)blockIdx.x * blockDim.x + threadIdx.x;
  const size_t total = (size_t)BB * SS * SS;
  if (i >= total) return;
  int b   = (int)(i / ((size_t)SS * SS));
  int rem = (int)(i % ((size_t)SS * SS));
  int r = rem / SS, c = rem % SS;
  const float* a = sp + ((size_t)b * SS + r) * 3;
  const float* q = sp + ((size_t)b * SS + c) * 3;
  float d0 = a[0] - q[0], d1 = a[1] - q[1], d2 = a[2] - q[2];
  dist[i] = sqrtf(d0 * d0 + d1 * d1 + d2 * d2 + 1e-12f);
}

__global__ void ln_bf16_kernel(const float* __restrict__ x, const float* __restrict__ g,
                               const float* __restrict__ b, bf16_t* __restrict__ out) {
  __shared__ float red[256];
  const int row = blockIdx.x, tid = threadIdx.x;
  const float* xr = x + (size_t)row * DD;
  float s = 0.f;
  for (int i = tid; i < DD; i += 256) s += xr[i];
  red[tid] = s; __syncthreads();
  for (int w = 128; w > 0; w >>= 1) { if (tid < w) red[tid] += red[tid + w]; __syncthreads(); }
  float mean = red[0] / (float)DD; __syncthreads();
  float v = 0.f;
  for (int i = tid; i < DD; i += 256) { float d = xr[i] - mean; v += d * d; }
  red[tid] = v; __syncthreads();
  for (int w = 128; w > 0; w >>= 1) { if (tid < w) red[tid] += red[tid + w]; __syncthreads(); }
  float rstd = rsqrtf(red[0] / (float)DD + 1e-5f);
  bf16_t* orow = out + (size_t)row * DD;
  for (int i = tid; i < DD; i += 256)
    orow[i] = (bf16_t)((xr[i] - mean) * rstd * g[i] + b[i]);
}

// ------------------------------------------------------------------
// generic WMMA GEMM: C = act(A @ B + bias) + res
//   A: (M x K) bf16 row-major; B given TRANSPOSED: (N x K) bf16 row-major
//   one wave per 32x64 tile; 2x4 fragments; K-loop manually unrolled x2
//   (16 WMMAs + 12 b128 fragment loads per iteration, K % 64 == 0)
// ------------------------------------------------------------------
template <bool HAS_BIAS, bool HAS_RES, bool DO_GELU, bool OUT_F32, bool OUT_BF16>
__global__ void __launch_bounds__(32)
__attribute__((amdgpu_waves_per_eu(1, 4)))
gemm_bf16_wmma(const bf16_t* __restrict__ A, int lda,
               const bf16_t* __restrict__ Bt, int ldbt,
               const float* __restrict__ bias,
               const float* __restrict__ res, int ldres,
               float* __restrict__ Cf, int ldc,
               bf16_t* __restrict__ Cb, int ldcb,
               int K) {
  const int n0 = blockIdx.x * 64;
  const int m0 = blockIdx.y * 32;
  v8f c[2][4];
#pragma unroll
  for (int mi = 0; mi < 2; ++mi)
#pragma unroll
    for (int ni = 0; ni < 4; ++ni) c[mi][ni] = zero_v8f();

  const bf16_t* a0p = A + (size_t)m0 * lda;
  const bf16_t* a1p = A + (size_t)(m0 + 16) * lda;
  for (int k0 = 0; k0 < K; k0 += 64) {
    // speculative global_prefetch of the next K slab (drops silently at end)
    __builtin_prefetch(a0p + k0 + 64, 0, 3);
    __builtin_prefetch(a1p + k0 + 64, 0, 3);
#pragma unroll
    for (int kk = 0; kk < 2; ++kk) {
      const int k = k0 + kk * 32;
      v16bf a0 = load_a_frag(a0p + k, lda);
      v16bf a1 = load_a_frag(a1p + k, lda);
#pragma unroll
      for (int ni = 0; ni < 4; ++ni) {
        v16bf b = load_bt_frag(Bt + (size_t)(n0 + ni * 16) * ldbt + k, ldbt);
        c[0][ni] = wmma_bf16(a0, b, c[0][ni]);
        c[1][ni] = wmma_bf16(a1, b, c[1][ni]);
      }
    }
  }

  const int lane = threadIdx.x & 31;
  const int col  = lane & 15;
  const int rb   = (lane < 16) ? 0 : 8;
#pragma unroll
  for (int mi = 0; mi < 2; ++mi) {
#pragma unroll
    for (int ni = 0; ni < 4; ++ni) {
      const int gc = n0 + ni * 16 + col;
#pragma unroll
      for (int r = 0; r < 8; ++r) {
        const int gr = m0 + mi * 16 + rb + r;
        float v = c[mi][ni][r];
        if (HAS_BIAS) v += bias[gc];
        if (DO_GELU)  v = 0.5f * v * (1.0f + erff(v * 0.70710678118654752f));
        if (HAS_RES)  v += res[(size_t)gr * ldres + gc];
        if (OUT_F32)  Cf[(size_t)gr * ldc  + gc] = v;
        if (OUT_BF16) Cb[(size_t)gr * ldcb + gc] = (bf16_t)v;
      }
    }
  }
}

// ------------------------------------------------------------------
// fused flash-attention with distance penalty + de/fe statistics
// grid: (T/16, H, B), block: 32 (one wave, EXEC all-ones around WMMA)
// qkv: (B*T x 3D) bf16 (Q,K slices); vT: dh-major V (b,h,dh,t) bf16;
// dist: (B,T,T) f32; o_out: (B*T x D) bf16
// Online softmax: only the row-max is reduced in-loop (4 DPP ops);
// l / w*dist / w*s partials are kept lane-local (the rescale factor is
// row-uniform) and reduced once at the end.
// ------------------------------------------------------------------
__global__ void __launch_bounds__(32)
__attribute__((amdgpu_waves_per_eu(1, 4)))
attn_wmma_kernel(const bf16_t* __restrict__ qkv,
                 const bf16_t* __restrict__ vT,
                 const float* __restrict__ dist,
                 bf16_t* __restrict__ o_out,
                 float* __restrict__ acc) {
  __shared__ bf16_t pbuf[16 * 32];
  const int qt = blockIdx.x, h = blockIdx.y, b = blockIdx.z;
  const int lane = threadIdx.x & 31;
  const int col  = lane & 15;
  const int rb   = (lane < 16) ? 0 : 8;
  const int ld   = D3;

  const bf16_t* qbase = qkv + ((size_t)(b * SS + qt * 16)) * ld + h * DHD;
  const bf16_t* kbase = qkv + ((size_t)b * SS) * ld + DD + h * DHD;
  const bf16_t* vtb   = vT  + ((size_t)(b * HH + h) * DHD) * SS;   // (dh x T) row-major

  const v16bf qa0 = load_a_frag(qbase,      ld);   // K 0..31 of dh
  const v16bf qa1 = load_a_frag(qbase + 32, ld);   // K 32..63 of dh

  float mrun[8], lsum[8], dsum[8], tsum[8];
  v8f o0 = zero_v8f(), o1 = zero_v8f(), o2 = zero_v8f(), o3 = zero_v8f();
#pragma unroll
  for (int r = 0; r < 8; ++r) { mrun[r] = -3.0e38f; lsum[r] = 0.f; dsum[r] = 0.f; tsum[r] = 0.f; }

  const float scale = 0.125f;                // 1/sqrt(64)
  const int rowmax = qt * 16 + 15;

  for (int j0 = 0; j0 <= rowmax; j0 += 32) {
    // S tiles: cols [j0, j0+16) and [j0+16, j0+32), K accumulated over dh=64
    v8f s0 = zero_v8f(), s1 = zero_v8f();
    s0 = wmma_bf16(qa0, load_bt_frag(kbase + (size_t)j0 * ld,             ld), s0);
    s0 = wmma_bf16(qa1, load_bt_frag(kbase + (size_t)j0 * ld + 32,        ld), s0);
    s1 = wmma_bf16(qa0, load_bt_frag(kbase + (size_t)(j0 + 16) * ld,      ld), s1);
    s1 = wmma_bf16(qa1, load_bt_frag(kbase + (size_t)(j0 + 16) * ld + 32, ld), s1);

#pragma unroll
    for (int r = 0; r < 8; ++r) {
      const int grow = qt * 16 + rb + r;
      const int gc0 = j0 + col;
      const int gc1 = j0 + 16 + col;
      const float* drow = dist + ((size_t)(b * SS + grow)) * SS;
      const float d0 = drow[gc0];
      const float d1 = drow[gc1];
      const bool msk0 = gc0 > grow;
      const bool msk1 = gc1 > grow;
      const float v0 = s0[r] * scale - 1.0f * d0;   // PEN = 1.0
      const float v1 = s1[r] * scale - 1.0f * d1;
      const float mx = rmax16(fmaxf(msk0 ? -3.0e38f : v0, msk1 ? -3.0e38f : v1));
      const float mnew = fmaxf(mrun[r], mx);
      const float corr = __expf(mrun[r] - mnew);    // row-uniform across 16 lanes
      const float e0 = msk0 ? 0.f : __expf(v0 - mnew);
      const float e1 = msk1 ? 0.f : __expf(v1 - mnew);
      lsum[r] = lsum[r] * corr + (e0 + e1);
      dsum[r] = dsum[r] * corr + (e0 * d0 + e1 * d1);
      tsum[r] = tsum[r] * corr + (msk0 ? 0.f : e0 * v0) + (msk1 ? 0.f : e1 * v1);
      mrun[r] = mnew;
      o0[r] *= corr; o1[r] *= corr; o2[r] *= corr; o3[r] *= corr;
      // stage P row immediately (no p0/p1 register arrays)
      const int row = rb + r;
      pbuf[row * 32 + col]      = (bf16_t)e0;
      pbuf[row * 32 + 16 + col] = (bf16_t)e1;
    }

    asm volatile("s_wait_dscnt 0" ::: "memory");
    v16bf pa;
    {
      const int m  = lane & 15;
      const int kb = (lane < 16) ? 0 : 8;
      const bf16_t* rp = &pbuf[m * 32 + kb];
      v8bf lo = *reinterpret_cast<const v8bf*>(rp);
      v8bf hi = *reinterpret_cast<const v8bf*>(rp + 16);
      pa = pack16(lo, hi);
    }

    // O += P @ V  (V is dh-major: B fragments are contiguous b128 loads)
    o0 = wmma_bf16(pa, load_bt_frag(vtb + (size_t) 0 * SS + j0, SS), o0);
    o1 = wmma_bf16(pa, load_bt_frag(vtb + (size_t)16 * SS + j0, SS), o1);
    o2 = wmma_bf16(pa, load_bt_frag(vtb + (size_t)32 * SS + j0, SS), o2);
    o3 = wmma_bf16(pa, load_bt_frag(vtb + (size_t)48 * SS + j0, SS), o3);
  }

  const float invBHT = 1.0f / (float)(BB * HH * SS);
#pragma unroll
  for (int r = 0; r < 8; ++r) {
    const int grow = qt * 16 + rb + r;
    const float l  = rsum16(lsum[r]);
    const float dw = rsum16(dsum[r]);
    const float tw = rsum16(tsum[r]);
    const float inv = 1.0f / l;
    bf16_t* orow = o_out + ((size_t)(b * SS + grow)) * DD + h * DHD;
    orow[ 0 + col] = (bf16_t)(o0[r] * inv);
    orow[16 + col] = (bf16_t)(o1[r] * inv);
    orow[32 + col] = (bf16_t)(o2[r] * inv);
    orow[48 + col] = (bf16_t)(o3[r] * inv);
    if (col == 0) {
      atomicAdd(acc + 1, dw * inv * invBHT);
      atomicAdd(acc + 2, (mrun[r] + __logf(l) - tw * inv) * invBHT);
    }
  }
}

// ------------------------------------------------------------------
// cross-entropy over V=256 logits per row (block per row, thread per logit)
// ------------------------------------------------------------------
__global__ void ce_kernel(const float* __restrict__ logits, const int* __restrict__ targets,
                          float* __restrict__ acc) {
  __shared__ float red[256];
  const int row = blockIdx.x, tid = threadIdx.x;
  const float v = logits[(size_t)row * VV + tid];
  red[tid] = v; __syncthreads();
  for (int w = 128; w > 0; w >>= 1) { if (tid < w) red[tid] = fmaxf(red[tid], red[tid + w]); __syncthreads(); }
  const float mx = red[0]; __syncthreads();
  red[tid] = __expf(v - mx); __syncthreads();
  for (int w = 128; w > 0; w >>= 1) { if (tid < w) red[tid] += red[tid + w]; __syncthreads(); }
  const float lse = mx + __logf(red[0]);
  if (tid == targets[row]) atomicAdd(acc + 0, (lse - v) * (1.0f / (float)(BB * SS)));
}

__global__ void rep_kernel(const float* __restrict__ tp, float* __restrict__ acc) {
  __shared__ float red[256];
  const int i = blockIdx.x, j = threadIdx.x;
  float r = 0.f;
  if (i != j) {
    float dx = tp[i * 3 + 0] - tp[j * 3 + 0];
    float dy = tp[i * 3 + 1] - tp[j * 3 + 1];
    float dz = tp[i * 3 + 2] - tp[j * 3 + 2];
    float d = sqrtf(dx * dx + dy * dy + dz * dz + 1e-12f);
    r = 1.0f / (d + 1e-4f);
  }
  red[j] = r; __syncthreads();
  for (int w = 128; w > 0; w >>= 1) { if (j < w) red[j] += red[j + w]; __syncthreads(); }
  if (j == 0) atomicAdd(acc + 3, red[0]);
}

__global__ void finalize_kernel(const float* __restrict__ acc, float* __restrict__ out) {
  const float ce  = acc[0];
  const float de  = acc[1];
  const float fe  = acc[2];
  const float rep = acc[3] / (float)(VV * (VV - 1));
  out[0] = ce + 0.01f * de + 0.01f * fe + 0.01f * rep;  // loss
  out[1] = de;
  out[2] = fe;
  out[3] = rep;
}

// ------------------------------------------------------------------
// host driver
// ------------------------------------------------------------------
extern "C" void kernel_launch(void* const* d_in, const int* in_sizes, int n_in,
                              void* d_out, int out_size, void* d_ws, size_t ws_size,
                              hipStream_t stream) {
  (void)in_sizes; (void)n_in; (void)out_size; (void)ws_size;

  const int*   idx       = (const int*)  d_in[0];
  const int*   targets   = (const int*)  d_in[1];
  const float* token_emb = (const float*)d_in[2];
  const float* pos_emb   = (const float*)d_in[3];
  const float* tok_pos   = (const float*)d_in[4];
  const float* ppw       = (const float*)d_in[5];
  const float* ln1_g     = (const float*)d_in[6];
  const float* ln1_b     = (const float*)d_in[7];
  const float* Wqkv      = (const float*)d_in[8];
  const float* Wproj     = (const float*)d_in[9];
  const float* ln2_g     = (const float*)d_in[10];
  const float* ln2_b     = (const float*)d_in[11];
  const float* Wff1      = (const float*)d_in[12];
  const float* bff1      = (const float*)d_in[13];
  const float* Wff2      = (const float*)d_in[14];
  const float* bff2      = (const float*)d_in[15];
  const float* lnf_g     = (const float*)d_in[16];
  const float* lnf_b     = (const float*)d_in[17];

  char* ws = (char*)d_ws;
  size_t off = 0;
  auto walloc = [&](size_t bytes) -> char* {
    char* p = ws + off;
    off = (off + bytes + 255) & ~(size_t)255;
    return p;
  };

  float*  x      = (float*) walloc((size_t)BT * DD * 4);
  float*  seqpos = (float*) walloc((size_t)BT * 3  * 4);
  float*  dist   = (float*) walloc((size_t)BB * SS * SS * 4);
  bf16_t* hb     = (bf16_t*)walloc((size_t)BT * DD * 2);
  bf16_t* qkvb   = (bf16_t*)walloc((size_t)BT * D3 * 2);
  bf16_t* vT     = (bf16_t*)walloc((size_t)BB * DD * SS * 2);
  bf16_t* ob     = (bf16_t*)walloc((size_t)BT * DD * 2);
  bf16_t* gb     = (bf16_t*)walloc((size_t)BT * D4 * 2);
  bf16_t* teb    = (bf16_t*)walloc((size_t)VV * DD * 2);
  bf16_t* WqkvT  = (bf16_t*)walloc((size_t)LL * DD * D3 * 2);
  bf16_t* WprojT = (bf16_t*)walloc((size_t)LL * DD * DD * 2);
  bf16_t* Wff1T  = (bf16_t*)walloc((size_t)LL * DD * D4 * 2);
  bf16_t* Wff2T  = (bf16_t*)walloc((size_t)LL * D4 * DD * 2);
  float*  acc    = (float*) walloc(64);

  zero_acc_kernel<<<1, 32, 0, stream>>>(acc);

  // weight precision/layout conversion (f32 -> bf16, B operands transposed to N x K)
  cvt_bf16_kernel<<<((size_t)VV * DD + 255) / 256, 256, 0, stream>>>(token_emb, teb, (size_t)VV * DD);
  cvt_t_bf16_kernel<<<dim3((DD * D3 + 255) / 256, 1, LL), 256, 0, stream>>>(Wqkv,  WqkvT,  DD, D3);
  cvt_t_bf16_kernel<<<dim3((DD * DD + 255) / 256, 1, LL), 256, 0, stream>>>(Wproj, WprojT, DD, DD);
  cvt_t_bf16_kernel<<<dim3((DD * D4 + 255) / 256, 1, LL), 256, 0, stream>>>(Wff1,  Wff1T,  DD, D4);
  cvt_t_bf16_kernel<<<dim3((D4 * DD + 255) / 256, 1, LL), 256, 0, stream>>>(Wff2,  Wff2T,  D4, DD);

  embed_kernel<<<BT, 256, 0, stream>>>(idx, token_emb, pos_emb, tok_pos, ppw, x, seqpos);
  dist_kernel<<<((size_t)BB * SS * SS + 255) / 256, 256, 0, stream>>>(seqpos, dist);

  for (int l = 0; l < LL; ++l) {
    // h = ln1(x) -> bf16
    ln_bf16_kernel<<<BT, 256, 0, stream>>>(x, ln1_g + l * DD, ln1_b + l * DD, hb);
    // qkv = h @ Wqkv (bf16 out)
    gemm_bf16_wmma<false, false, false, false, true>
        <<<dim3(D3 / 64, BT / 32), 32, 0, stream>>>(
        hb, DD, WqkvT + (size_t)l * DD * D3, DD,
        nullptr, nullptr, 0, nullptr, 0, qkvb, D3, DD);
    // repack V to dh-major for contiguous B-fragment loads
    vtrans_kernel<<<((size_t)BB * DD * SS + 255) / 256, 256, 0, stream>>>(qkvb, vT);
    // fused attention (with distance penalty, causal mask, de/fe stats)
    attn_wmma_kernel<<<dim3(SS / 16, HH, BB), 32, 0, stream>>>(qkvb, vT, dist, ob, acc);
    // x = x + o @ Wproj (f32 out, in-place residual)
    gemm_bf16_wmma<false, true, false, true, false>
        <<<dim3(DD / 64, BT / 32), 32, 0, stream>>>(
        ob, DD, WprojT + (size_t)l * DD * DD, DD,
        nullptr, x, DD, x, DD, nullptr, 0, DD);
    // h = ln2(x) -> bf16
    ln_bf16_kernel<<<BT, 256, 0, stream>>>(x, ln2_g + l * DD, ln2_b + l * DD, hb);
    // g = gelu(h @ W1 + b1) (bf16 out)
    gemm_bf16_wmma<true, false, true, false, true>
        <<<dim3(D4 / 64, BT / 32), 32, 0, stream>>>(
        hb, DD, Wff1T + (size_t)l * DD * D4, DD,
        bff1 + l * D4, nullptr, 0, nullptr, 0, gb, D4, DD);
    // x = x + g @ W2 + b2
    gemm_bf16_wmma<true, true, false, true, false>
        <<<dim3(DD / 64, BT / 32), 32, 0, stream>>>(
        gb, D4, Wff2T + (size_t)l * D4 * DD, D4,
        bff2 + l * DD, x, DD, x, DD, nullptr, 0, D4);
  }

  // final LN + tied lm_head (token_emb is already (V x D) = transposed-B layout)
  ln_bf16_kernel<<<BT, 256, 0, stream>>>(x, lnf_g, lnf_b, hb);
  float* logits = (float*)d_out;
  gemm_bf16_wmma<false, false, false, true, false>
      <<<dim3(VV / 64, BT / 32), 32, 0, stream>>>(
      hb, DD, teb, DD, nullptr, nullptr, 0, logits, VV, nullptr, 0, DD);

  ce_kernel<<<BT, 256, 0, stream>>>(logits, targets, acc);
  rep_kernel<<<VV, VV, 0, stream>>>(tok_pos, acc);
  finalize_kernel<<<1, 1, 0, stream>>>(acc, logits + (size_t)BT * VV);
}
